// trajectory2seq_59562606461598
// MI455X (gfx1250) — matrix operand order repaired and checked by reading, not compiled, not run on hardware
//
#include <hip/hip_runtime.h>
#include <cstdint>

#define B_ 256
#define S_ 256
#define H_ 512
#define D_ 32
#define T_ 32
#define L_ 2

typedef __attribute__((ext_vector_type(16))) __bf16 v16bf;
typedef __attribute__((ext_vector_type(8)))  float  v8f;

static __device__ __forceinline__ unsigned short f2bf(float f) {
  return (unsigned short)(__float_as_uint(f) >> 16);   // truncate fp32 -> bf16
}
static __device__ __forceinline__ float bf2f(unsigned short s) {
  return __uint_as_float(((unsigned)s) << 16);
}

union PFrag { uint4 q; v16bf v; };

// ---------------------------------------------------------------------------
// Packed fragment blobs.
// Both A (16x32, M-major tiles) and B (32x16, N-major tiles) fragments are
// stored as [tile][lane][16 halves] so a wave fetches one fragment with a
// single per-lane global_load_b128 and zero repacking VALU.
//   A element (m,k): lane = (m&15) + (k&8 ? 16:0), half = (k&7) + (k&16 ? 8:0)
//       tile = (m>>4)*Kt + (k>>5)
//   B element (k,n) = W[n,k]: lane = (n&15) + (k&8 ? 16:0), half = (k&7) + (k&16 ? 8:0)
//       tile = (n>>4)*Kt + (k>>5)
// (matches ISA 7.12.2 16-bit A 16x32 / B 32x16 VGPR layouts)
// ---------------------------------------------------------------------------
static __device__ __forceinline__ v16bf load_frag_packed(
    const unsigned short* __restrict__ P, size_t tile, int lane) {
  PFrag u;
  u.q = *(const uint4*)(P + (tile * 32 + (size_t)lane) * 16);
  return u.v;
}

static __device__ __forceinline__ void store_packed_a(
    unsigned short* __restrict__ P, int Kt, int m, int k, unsigned short val) {
  const size_t tile = (size_t)(m >> 4) * Kt + (k >> 5);
  const int lane = (m & 15) + ((k & 8) ? 16 : 0);
  const int half = (k & 7) + ((k & 16) ? 8 : 0);
  P[(tile * 32 + lane) * 16 + half] = val;
}

static __device__ __forceinline__ v8f wmma_bf16(v16bf a, v16bf b, v8f c) {
  return __builtin_amdgcn_wmma_f32_16x16x32_bf16(false, a, false, b, (short)0, c, false, false);
}

static __device__ __forceinline__ float sigmoidf_(float x) {
  return 1.0f / (1.0f + __expf(-x));
}

// ---------------------------------------------------------------------------
// One-time weight pre-swizzle: f32 [N,K] row-major -> packed bf16 B fragments.
// grid = (N/16, K/32), block = 32 (one wave per 16x32 tile).
// ---------------------------------------------------------------------------
__global__ __launch_bounds__(32) void pack_b_kernel(
    const float* __restrict__ W, int K, unsigned short* __restrict__ out) {
  const int lane = threadIdx.x;
  const int n0 = blockIdx.x * 16;
  const int k0 = blockIdx.y * 32;
  const float* row = W + (size_t)(n0 + (lane & 15)) * K + k0 + ((lane & 16) ? 8 : 0);
  float4 p0 = *(const float4*)(row);
  float4 p1 = *(const float4*)(row + 4);
  float4 p2 = *(const float4*)(row + 16);
  float4 p3 = *(const float4*)(row + 20);
  const float f[16] = {p0.x, p0.y, p0.z, p0.w, p1.x, p1.y, p1.z, p1.w,
                       p2.x, p2.y, p2.z, p2.w, p3.x, p3.y, p3.z, p3.w};
  unsigned short s[16];
#pragma unroll
  for (int j = 0; j < 16; ++j) s[j] = f2bf(f[j]);
  const size_t tile = (size_t)blockIdx.x * gridDim.y + blockIdx.y;
  *(uint4*)(out + (tile * 32 + (size_t)lane) * 16) = *(const uint4*)s;
}

// ---------------------------------------------------------------------------
// Fused GRU cell: h' = (1-z)*n + z*h. One wave = one 16x16 tile of h'.
// Templated on the input-size case so the hot K-loop is branch-free:
//   KIN512=true : 8 b128 loads + 6 WMMA per K=32 slice (x and h GEMMs fused)
//   KIN512=false: 4 b128 loads + 3 WMMA per slice; x (2 features) as scalars
// grid = (B/16, H/64), block = 128.
// ---------------------------------------------------------------------------
template <bool KIN512>
__global__ __launch_bounds__(128) void gru_cell_kernel(
    const unsigned short* __restrict__ xP,   // packed A activations (KIN512)
    const float* __restrict__ xF,            // f32 x (scalar path, !KIN512)
    long ldx,
    const unsigned short* __restrict__ hP,   // packed A prev hidden
    const float* __restrict__ hF,            // f32 prev hidden (for z*h)
    const unsigned short* __restrict__ WihP, // packed B [3H,512] (KIN512)
    const float* __restrict__ Wih2,          // f32 Wih [3H,2] (!KIN512)
    const unsigned short* __restrict__ WhhP, // packed B [3H,H]
    const float* __restrict__ bih,
    const float* __restrict__ bhh,
    float* __restrict__ houtF,               // f32 [B,H]
    unsigned short* __restrict__ houtP,      // packed A [B,H]
    unsigned short* __restrict__ side,       // optional bf16 enc cache [B,S,H]
    long side_stride)
{
  const int lane = threadIdx.x & 31;
  const int wave = threadIdx.x >> 5;
  const int m0 = blockIdx.x * 16;
  const int ntile = blockIdx.y * 4 + wave;   // output column tile, n0 = ntile*16
  const int n0 = ntile * 16;
  const int KT = H_ / 32;                    // 16 K-tiles

  v8f acc_r = {};
  v8f acc_z = {};
  v8f acc_in = {};
  v8f acc_hn = {};

  const size_t atile0 = (size_t)(m0 >> 4) * KT;
  const size_t br_t = (size_t)ntile * KT;               // gate r row block
  const size_t bz_t = (size_t)(ntile + 32) * KT;        // gate z (rows + H)
  const size_t bn_t = (size_t)(ntile + 64) * KT;        // gate n (rows + 2H)

  for (int kt = 0; kt < KT; ++kt) {
    v16bf ah  = load_frag_packed(hP, atile0 + kt, lane);
    v16bf bhr = load_frag_packed(WhhP, br_t + kt, lane);
    v16bf bhz = load_frag_packed(WhhP, bz_t + kt, lane);
    v16bf bhn = load_frag_packed(WhhP, bn_t + kt, lane);
    acc_r  = wmma_bf16(ah, bhr, acc_r);
    acc_z  = wmma_bf16(ah, bhz, acc_z);
    acc_hn = wmma_bf16(ah, bhn, acc_hn);
    if (KIN512) {
      v16bf ax  = load_frag_packed(xP, atile0 + kt, lane);
      v16bf bxr = load_frag_packed(WihP, br_t + kt, lane);
      v16bf bxz = load_frag_packed(WihP, bz_t + kt, lane);
      v16bf bxn = load_frag_packed(WihP, bn_t + kt, lane);
      acc_r  = wmma_bf16(ax, bxr, acc_r);
      acc_z  = wmma_bf16(ax, bxz, acc_z);
      acc_in = wmma_bf16(ax, bxn, acc_in);
    }
  }

  const int j = n0 + (lane & 15);
  const float br   = bih[j] + bhh[j];
  const float bz   = bih[j + H_] + bhh[j + H_];
  const float bin_ = bih[j + 2 * H_];
  const float bhn_ = bhh[j + 2 * H_];

  float wr0 = 0.f, wr1 = 0.f, wz0 = 0.f, wz1 = 0.f, wn0 = 0.f, wn1 = 0.f;
  if (!KIN512) {
    wr0 = Wih2[(size_t)j * 2];              wr1 = Wih2[(size_t)j * 2 + 1];
    wz0 = Wih2[(size_t)(j + H_) * 2];       wz1 = Wih2[(size_t)(j + H_) * 2 + 1];
    wn0 = Wih2[(size_t)(j + 2 * H_) * 2];   wn1 = Wih2[(size_t)(j + 2 * H_) * 2 + 1];
  }

#pragma unroll
  for (int i = 0; i < 8; ++i) {
    const int b = m0 + i + ((lane & 16) ? 8 : 0);
    float gr  = acc_r[i]  + br;
    float gz  = acc_z[i]  + bz;
    float gin = acc_in[i] + bin_;
    float ghn = acc_hn[i] + bhn_;
    if (!KIN512) {
      const float x0 = xF[(size_t)b * ldx];
      const float x1 = xF[(size_t)b * ldx + 1];
      gr  += x0 * wr0 + x1 * wr1;
      gz  += x0 * wz0 + x1 * wz1;
      gin += x0 * wn0 + x1 * wn1;
    }
    const float r = sigmoidf_(gr);
    const float z = sigmoidf_(gz);
    const float n = tanhf(gin + r * ghn);
    const float hold = hF[(size_t)b * H_ + j];
    const float hv = (1.0f - z) * n + z * hold;
    houtF[(size_t)b * H_ + j] = hv;
    const unsigned short hb = f2bf(hv);
    store_packed_a(houtP, KT, b, j, hb);
    if (side) side[(size_t)b * side_stride + j] = hb;
  }
}

// ---------------------------------------------------------------------------
// Generic WMMA linear: out[b,n] = sum_k X[b,k]*W[n,k] + bias[n]
// X and W pre-swizzled bf16; optional f32 and/or packed-A outputs.
// grid = (B/16, ceil(N/64)), block = 128.
// ---------------------------------------------------------------------------
__global__ __launch_bounds__(128) void linear_kernel(
    const unsigned short* __restrict__ XP, int K,
    const unsigned short* __restrict__ WP,
    const float* __restrict__ bias,
    float* __restrict__ outF,
    unsigned short* __restrict__ outP,
    int N)
{
  const int lane = threadIdx.x & 31;
  const int wave = threadIdx.x >> 5;
  const int m0 = blockIdx.x * 16;
  const int ntile = blockIdx.y * 4 + wave;
  if (ntile * 16 >= N) return;   // wave-uniform early exit
  const int KT = K / 32;

  v8f acc = {};
  const size_t at = (size_t)(m0 >> 4) * KT;
  const size_t bt = (size_t)ntile * KT;
  for (int kt = 0; kt < KT; ++kt) {
    v16bf a = load_frag_packed(XP, at + kt, lane);
    v16bf b = load_frag_packed(WP, bt + kt, lane);
    acc = wmma_bf16(a, b, acc);
  }
  const int j = ntile * 16 + (lane & 15);
  const float bv = bias[j];
#pragma unroll
  for (int i = 0; i < 8; ++i) {
    const int b = m0 + i + ((lane & 16) ? 8 : 0);
    const float v = acc[i] + bv;
    if (outF) outF[(size_t)b * N + j] = v;
    if (outP) store_packed_a(outP, N / 32, b, j, f2bf(v));
  }
}

// scores[b,s] = q[b,:] . enc[b,s,:]; grid = (B, S/8), block = 256
__global__ __launch_bounds__(256) void scores_kernel(
    const float* __restrict__ q,
    const unsigned short* __restrict__ enc,
    float* __restrict__ scores)
{
  const int b = blockIdx.x;
  const int lane = threadIdx.x & 31;
  const int s = blockIdx.y * 8 + (threadIdx.x >> 5);
  const unsigned short* er = enc + ((size_t)b * S_ + s) * H_;
  const float* qr = q + (size_t)b * H_;
  float acc = 0.f;
#pragma unroll 4
  for (int hh = lane; hh < H_; hh += 32) acc += qr[hh] * bf2f(er[hh]);
#pragma unroll
  for (int m = 16; m >= 1; m >>= 1) acc += __shfl_xor(acc, m, 32);
  if (lane == 0) scores[(size_t)b * S_ + s] = acc;
}

// softmax over s; writes w to wbuf and to attn output [B,S,T]
__global__ __launch_bounds__(256) void softmax_attn_kernel(
    const float* __restrict__ scores,
    float* __restrict__ wbuf,
    float* __restrict__ attn_out,
    int t)
{
  __shared__ float red[256];
  const int b = blockIdx.x;
  const int tid = threadIdx.x;
  const float v = scores[(size_t)b * S_ + tid];
  red[tid] = v;
  __syncthreads();
  for (int s = 128; s > 0; s >>= 1) {
    if (tid < s) red[tid] = fmaxf(red[tid], red[tid + s]);
    __syncthreads();
  }
  const float mx = red[0];
  __syncthreads();
  const float e = __expf(v - mx);
  red[tid] = e;
  __syncthreads();
  for (int s = 128; s > 0; s >>= 1) {
    if (tid < s) red[tid] += red[tid + s];
    __syncthreads();
  }
  const float w = e / red[0];
  wbuf[(size_t)b * S_ + tid] = w;
  attn_out[(size_t)b * S_ * T_ + (size_t)tid * T_ + t] = w;
}

// att[b,h] = sum_s w[b,s]*enc[b,s,h]; grid = (B, H/256), block = 256
__global__ __launch_bounds__(256) void attvec_kernel(
    const float* __restrict__ wbuf,
    const unsigned short* __restrict__ enc,
    float* __restrict__ att)
{
  const int b = blockIdx.x;
  const int hh = blockIdx.y * 256 + threadIdx.x;
  const float* wr = wbuf + (size_t)b * S_;
  const unsigned short* eb = enc + (size_t)b * S_ * H_ + hh;
  float acc = 0.f;
#pragma unroll 4
  for (int s = 0; s < S_; ++s) acc += wr[s] * bf2f(eb[(size_t)s * H_]);
  att[(size_t)b * H_ + hh] = acc;
}

// catP[b,:] = pack([d1[b,:], att[b,:]]) as A fragments (K=1024); grid = B, block = 512
__global__ __launch_bounds__(512) void cat_kernel(
    const float* __restrict__ d1, const float* __restrict__ att,
    unsigned short* __restrict__ catP)
{
  const int b = blockIdx.x;
  const int tid = threadIdx.x;
  store_packed_a(catP, 2 * H_ / 32, b, tid,       f2bf(d1[(size_t)b * H_ + tid]));
  store_packed_a(catP, 2 * H_ / 32, b, H_ + tid,  f2bf(att[(size_t)b * H_ + tid]));
}

// argmax over D=32 (one wave per b) + logits -> vec_out[b,t,:]
__global__ __launch_bounds__(32) void argmax_logits_kernel(
    const float* __restrict__ logits,
    int* __restrict__ tok,
    float* __restrict__ vec_out,
    int t)
{
  const int b = blockIdx.x;
  const int d = threadIdx.x;
  const float lv = logits[(size_t)b * D_ + d];
  vec_out[(size_t)b * T_ * D_ + (size_t)t * D_ + d] = lv;
  float v = lv;
  int idx = d;
#pragma unroll
  for (int off = 16; off >= 1; off >>= 1) {
    const float ov = __shfl_xor(v, off, 32);
    const int oi = __shfl_xor(idx, off, 32);
    if (ov > v || (ov == v && oi < idx)) { v = ov; idx = oi; }   // first-max tie break
  }
  if (d == 0) tok[b] = idx;
}

// eP[b,:] = pack(emb[tok[b],:]); grid = B, block = 256
__global__ __launch_bounds__(256) void emb_gather_kernel(
    const int* __restrict__ tok, const float* __restrict__ emb,
    unsigned short* __restrict__ eP)
{
  const int b = blockIdx.x;
  const int tid = threadIdx.x;
  const int tk = tok[b];
  store_packed_a(eP, H_ / 32, b, tid,       f2bf(emb[(size_t)tk * H_ + tid]));
  store_packed_a(eP, H_ / 32, b, 256 + tid, f2bf(emb[(size_t)tk * H_ + 256 + tid]));
}

__global__ __launch_bounds__(256) void init_kernel(
    float* h0, float* h1, unsigned short* h0p, unsigned short* h1p, int* tok) {
  const int i = blockIdx.x * 256 + threadIdx.x;
  if (i < B_ * H_) {
    h0[i] = 0.f; h1[i] = 0.f;
    h0p[i] = 0;  h1p[i] = 0;
  }
  if (i < B_) tok[i] = 0;
}

__global__ __launch_bounds__(256) void hidden_copy_kernel(
    const float* __restrict__ d0, const float* __restrict__ d1,
    float* __restrict__ out)
{
  const int i = blockIdx.x * 256 + threadIdx.x;
  if (i < B_ * H_) {
    out[i] = d0[i];
    out[B_ * H_ + i] = d1[i];
  }
}

// ---------------------------------------------------------------------------
extern "C" void kernel_launch(void* const* d_in, const int* in_sizes, int n_in,
                              void* d_out, int out_size, void* d_ws, size_t ws_size,
                              hipStream_t stream) {
  (void)in_sizes; (void)n_in; (void)out_size; (void)ws_size;

  const float* x     = (const float*)d_in[0];
  const float* eWih0 = (const float*)d_in[1];
  const float* eWhh0 = (const float*)d_in[2];
  const float* ebih0 = (const float*)d_in[3];
  const float* ebhh0 = (const float*)d_in[4];
  const float* eWih1 = (const float*)d_in[5];
  const float* eWhh1 = (const float*)d_in[6];
  const float* ebih1 = (const float*)d_in[7];
  const float* ebhh1 = (const float*)d_in[8];
  const float* dWih  = (const float*)d_in[9];
  const float* dWhh  = (const float*)d_in[10];
  const float* dbih  = (const float*)d_in[11];
  const float* dbhh  = (const float*)d_in[12];
  const float* emb   = (const float*)d_in[13];
  const float* Wq    = (const float*)d_in[14];
  const float* bq    = (const float*)d_in[15];
  const float* Wc    = (const float*)d_in[16];
  const float* bc    = (const float*)d_in[17];
  const float* Wfc   = (const float*)d_in[18];
  const float* bfc   = (const float*)d_in[19];

  float* out      = (float*)d_out;
  float* vec_out  = out;                                 // [B,T,D]
  float* hid_out  = out + (size_t)B_ * T_ * D_;          // [L,B,H]
  float* attn_out = hid_out + (size_t)L_ * B_ * H_;      // [B,S,T]

  // ---- workspace carve-up ----
  char* wsp = (char*)d_ws;
  size_t off = 0;
  auto take = [&](size_t bytes) -> char* {
    char* p = wsp + off;
    off = (off + bytes + 255) & ~(size_t)255;
    return p;
  };
  typedef unsigned short u16;
  u16* enc_bf = (u16*)take((size_t)B_ * S_ * H_ * 2);    // 64 MB bf16 enc cache
  // packed bf16 weights (one-time per launch)
  const size_t wsz3 = (size_t)3 * H_ * H_ * 2;           // [3H,512] packed
  u16* eWhh0P = (u16*)take(wsz3);
  u16* eWih1P = (u16*)take(wsz3);
  u16* eWhh1P = (u16*)take(wsz3);
  u16* dWih0P = (u16*)take(wsz3);
  u16* dWhh0P = (u16*)take(wsz3);
  u16* dWih1P = (u16*)take(wsz3);
  u16* dWhh1P = (u16*)take(wsz3);
  u16* WqP  = (u16*)take((size_t)H_ * H_ * 2);
  u16* WcP  = (u16*)take((size_t)H_ * 2 * H_ * 2);
  u16* WfcP = (u16*)take((size_t)D_ * H_ * 2);
  // f32 state ping-pong + packed-A mirrors
  float* h0b[2] = { (float*)take((size_t)B_ * H_ * 4), (float*)take((size_t)B_ * H_ * 4) };
  float* h1b[2] = { (float*)take((size_t)B_ * H_ * 4), (float*)take((size_t)B_ * H_ * 4) };
  u16* h0p[2] = { (u16*)take((size_t)B_ * H_ * 2), (u16*)take((size_t)B_ * H_ * 2) };
  u16* h1p[2] = { (u16*)take((size_t)B_ * H_ * 2), (u16*)take((size_t)B_ * H_ * 2) };
  u16*   eP     = (u16*)take((size_t)B_ * H_ * 2);
  float* qbuf   = (float*)take((size_t)B_ * H_ * 4);
  float* scbuf  = (float*)take((size_t)B_ * S_ * 4);
  float* wbuf   = (float*)take((size_t)B_ * S_ * 4);
  float* attbuf = (float*)take((size_t)B_ * H_ * 4);
  u16*   catP   = (u16*)take((size_t)B_ * 2 * H_ * 2);
  u16*   outcP  = (u16*)take((size_t)B_ * H_ * 2);
  float* logits = (float*)take((size_t)B_ * D_ * 4);
  int*   tok    = (int*)take((size_t)B_ * 4);

  // ---- one-time weight pre-swizzle ----
  const dim3 pk32(32);
  pack_b_kernel<<<dim3(3 * H_ / 16, H_ / 32), pk32, 0, stream>>>(eWhh0, H_, eWhh0P);
  pack_b_kernel<<<dim3(3 * H_ / 16, H_ / 32), pk32, 0, stream>>>(eWih1, H_, eWih1P);
  pack_b_kernel<<<dim3(3 * H_ / 16, H_ / 32), pk32, 0, stream>>>(eWhh1, H_, eWhh1P);
  pack_b_kernel<<<dim3(3 * H_ / 16, H_ / 32), pk32, 0, stream>>>(dWih, H_, dWih0P);
  pack_b_kernel<<<dim3(3 * H_ / 16, H_ / 32), pk32, 0, stream>>>(dWhh, H_, dWhh0P);
  pack_b_kernel<<<dim3(3 * H_ / 16, H_ / 32), pk32, 0, stream>>>(dWih + (size_t)3 * H_ * H_, H_, dWih1P);
  pack_b_kernel<<<dim3(3 * H_ / 16, H_ / 32), pk32, 0, stream>>>(dWhh + (size_t)3 * H_ * H_, H_, dWhh1P);
  pack_b_kernel<<<dim3(H_ / 16, H_ / 32), pk32, 0, stream>>>(Wq, H_, WqP);
  pack_b_kernel<<<dim3(H_ / 16, 2 * H_ / 32), pk32, 0, stream>>>(Wc, 2 * H_, WcP);
  pack_b_kernel<<<dim3(D_ / 16, H_ / 32), pk32, 0, stream>>>(Wfc, H_, WfcP);

  const dim3 cellGrid(B_ / 16, H_ / 64);
  const dim3 cellBlk(128);

  init_kernel<<<dim3((B_ * H_ + 255) / 256), dim3(256), 0, stream>>>(
      h0b[0], h1b[0], h0p[0], h1p[0], tok);

  // ---- encoder: 256 steps, 2 layers ----
  for (int t = 0; t < S_; ++t) {
    const int p = t & 1;
    gru_cell_kernel<false><<<cellGrid, cellBlk, 0, stream>>>(
        nullptr, x + (size_t)t * 2, (long)(S_ * 2),
        h0p[p], h0b[p], nullptr, eWih0, eWhh0P, ebih0, ebhh0,
        h0b[p ^ 1], h0p[p ^ 1], nullptr, 0);
    gru_cell_kernel<true><<<cellGrid, cellBlk, 0, stream>>>(
        h0p[p ^ 1], nullptr, 0,
        h1p[p], h1b[p], eWih1P, nullptr, eWhh1P, ebih1, ebhh1,
        h1b[p ^ 1], h1p[p ^ 1], enc_bf + (size_t)t * H_, (long)(S_ * H_));
  }
  // final encoder states land in buffer 0 (256 even steps); decoder reuses
  // the same ping-pong buffers as its own state buffers.

  const float* dbih0 = dbih;
  const float* dbih1 = dbih + (size_t)3 * H_;
  const float* dbhh0 = dbhh;
  const float* dbhh1 = dbhh + (size_t)3 * H_;

  // ---- decoder: 32 steps ----
  for (int t = 0; t < T_; ++t) {
    const int p = t & 1;
    emb_gather_kernel<<<dim3(B_), dim3(256), 0, stream>>>(tok, emb, eP);
    gru_cell_kernel<true><<<cellGrid, cellBlk, 0, stream>>>(
        eP, nullptr, 0,
        h0p[p], h0b[p], dWih0P, nullptr, dWhh0P, dbih0, dbhh0,
        h0b[p ^ 1], h0p[p ^ 1], nullptr, 0);
    gru_cell_kernel<true><<<cellGrid, cellBlk, 0, stream>>>(
        h0p[p ^ 1], nullptr, 0,
        h1p[p], h1b[p], dWih1P, nullptr, dWhh1P, dbih1, dbhh1,
        h1b[p ^ 1], h1p[p ^ 1], nullptr, 0);
    linear_kernel<<<dim3(B_ / 16, 8), cellBlk, 0, stream>>>(
        h1p[p ^ 1], H_, WqP, bq, qbuf, nullptr, H_);
    scores_kernel<<<dim3(B_, S_ / 8), dim3(256), 0, stream>>>(qbuf, enc_bf, scbuf);
    softmax_attn_kernel<<<dim3(B_), dim3(256), 0, stream>>>(scbuf, wbuf, attn_out, t);
    attvec_kernel<<<dim3(B_, H_ / 256), dim3(256), 0, stream>>>(wbuf, enc_bf, attbuf);
    cat_kernel<<<dim3(B_), dim3(512), 0, stream>>>(h1b[p ^ 1], attbuf, catP);
    linear_kernel<<<dim3(B_ / 16, 8), cellBlk, 0, stream>>>(
        catP, 2 * H_, WcP, bc, nullptr, outcP, H_);
    linear_kernel<<<dim3(B_ / 16, 1), cellBlk, 0, stream>>>(
        outcP, H_, WfcP, bfc, logits, nullptr, D_);
    argmax_logits_kernel<<<dim3(B_), dim3(32), 0, stream>>>(logits, tok, vec_out, t);
  }

  // final decoder hidden states (t=31 odd -> written into buffer 0)
  hidden_copy_kernel<<<dim3((B_ * H_ + 255) / 256), dim3(256), 0, stream>>>(
      h0b[0], h1b[0], hid_out);
}
